// MultiHeadAttention_18150531793306
// MI455X (gfx1250) — matrix operand order repaired and checked by reading, not compiled, not run on hardware
//
#include <hip/hip_runtime.h>
#include <hip/hip_bf16.h>

// ---------------------------------------------------------------------------
// MHA forward for MI455X (gfx1250): wave32 + v_wmma_f32_16x16x32_f16,
// TDM (tensor_load_to_lds) + async global->LDS staging in the attention loop.
//   B=4, S=2048, D=768, H=12, DEPTH=64
// ---------------------------------------------------------------------------

#define BB   4
#define SS   2048
#define DD   768
#define HH   12
#define DEP  64

typedef __attribute__((ext_vector_type(16))) _Float16 v16h;
typedef __attribute__((ext_vector_type(8)))  _Float16 v8h;
typedef __attribute__((ext_vector_type(8)))  float    v8f;

typedef __attribute__((ext_vector_type(4))) unsigned int u32x4;
typedef __attribute__((ext_vector_type(8))) int          i32x8;
typedef __attribute__((ext_vector_type(4))) int          i32x4;
typedef __attribute__((ext_vector_type(4))) int          v4i;

#if __has_builtin(__builtin_amdgcn_tensor_load_to_lds)
#define HAVE_TDM 1
#else
#define HAVE_TDM 0
#endif
#if __has_builtin(__builtin_amdgcn_global_load_async_to_lds_b128)
#define HAVE_ASYNC 1
#else
#define HAVE_ASYNC 0
#endif

// Builtin wants v4i pointers in explicit address spaces:
//   param0: v4i addrspace(1)* (global source), param1: v4i addrspace(3)* (LDS)
typedef __attribute__((address_space(1))) v4i gv4i_t;
typedef __attribute__((address_space(3))) v4i lv4i_t;
__device__ __forceinline__ gv4i_t* as_global_v4i(const void* p) {
  return (gv4i_t*)(uintptr_t)p;
}
__device__ __forceinline__ lv4i_t* as_local_v4i(void* p) {
  return (lv4i_t*)(unsigned int)(uintptr_t)p;
}

// ---- Tensor Data Mover: 1-D tile of `nelem` f16 elements, global -> LDS ----
// D# built per CDNA5 ISA 8.3/8.4 (group0: count/lds/global/type, group1:
// data_size=2B, tensor_dim0=tile_dim0=nelem, stride=nelem, no multicast).
__device__ __forceinline__ void tdm_load_f16(const void* gsrc, void* lds_dst,
                                             unsigned int nelem) {
#if HAVE_TDM
  const unsigned long long ga = (unsigned long long)(uintptr_t)gsrc;
  const unsigned int       la = (unsigned int)(uintptr_t)lds_dst;
  u32x4 g0 = {};
  g0[0] = 1u;                                            // count=1 (valid)
  g0[1] = la;                                            // lds_addr (bytes)
  g0[2] = (unsigned int)ga;                              // global_addr[31:0]
  g0[3] = (unsigned int)((ga >> 32) & 0x01FFFFFFu)       // global_addr[56:32]
          | 0x80000000u;                                 // type=2 ("image")
  i32x8 g1 = {};
  g1[0] = (int)(1u << 16);                               // data_size=1 (2B)
  g1[1] = (int)((nelem & 0xFFFFu) << 16);                // tensor_dim0[15:0]
  g1[2] = (int)((nelem >> 16) & 0xFFFFu)                 // tensor_dim0[31:16]
          | (int)(1u << 16);                             // tensor_dim1 = 1
  g1[3] = (int)((nelem & 0xFFFFu) << 16);                // tile_dim0 = nelem
  g1[4] = 0;                                             // tile_dim1/2 unused
  g1[5] = (int)nelem;                                    // dim0_stride[31:0]
  g1[6] = 0;
  g1[7] = 0;
  const i32x4 gz = {};
#if defined(__clang_major__) && (__clang_major__ >= 23)
  const i32x8 gz8 = {};
  __builtin_amdgcn_tensor_load_to_lds(g0, g1, gz, gz, gz8, 0);
#else
  __builtin_amdgcn_tensor_load_to_lds(g0, g1, gz, gz, 0);
#endif
#else
  (void)gsrc; (void)lds_dst; (void)nelem;
#endif
}

// ---- WMMA fragment loaders (contiguous k order: lane half h holds k = 16h + s) ------------------
__device__ __forceinline__ v16h ld_a_h(const _Float16* p) {
  v16h r;
  const v8h lo = *(const v8h*)(p);
  const v8h hi = *(const v8h*)(p + 8);
#pragma unroll
  for (int h = 0; h < 8; ++h) { r[h] = lo[h]; r[h + 8] = hi[h]; }
  return r;
}
__device__ __forceinline__ v16h ld_a_f(const float* p) {
  v16h r;
#pragma unroll
  for (int h = 0; h < 8; ++h) {
    r[h]     = (_Float16)p[h];
    r[h + 8] = (_Float16)p[8 + h];
  }
  return r;
}
__device__ __forceinline__ v16h ld_b_h(const _Float16* p) {
  return *(const v16h*)(p);
}
__device__ __forceinline__ v16h ld_b_f(const float* p) {
  v16h r;
#pragma unroll
  for (int h = 0; h < 16; ++h) r[h] = (_Float16)p[h];
  return r;
}
// B fragment with per-half stride (row-major V in LDS: key stride = DEP)
__device__ __forceinline__ v16h ld_b_strided(const _Float16* p, int stride) {
  v16h r;
#pragma unroll
  for (int h = 0; h < 16; ++h) r[h] = p[h * stride];
  return r;
}

#define WMMA(A, Bf, C) \
  __builtin_amdgcn_wmma_f32_16x16x32_f16(false, (A), false, (Bf), (short)0, (C), false, false)

// ---------------------------------------------------------------------------
// Kernel 1: fused QKV projection.  Y = X @ W^T written as [B,H,S,DEP] f16.
// ---------------------------------------------------------------------------
__global__ void __launch_bounds__(128)
qkv_proj_kernel(const float* __restrict__ q, const float* __restrict__ k,
                const float* __restrict__ v, const float* __restrict__ wq,
                const float* __restrict__ wk, const float* __restrict__ wv,
                _Float16* __restrict__ qh, _Float16* __restrict__ kh,
                _Float16* __restrict__ vh) {
  const int wave = threadIdx.x >> 5;
  const int lane = threadIdx.x & 31;
  const int l16  = lane & 15;
  const int hi   = lane >> 4;

  const int z  = blockIdx.z;
  const float* X = (z == 0) ? q : (z == 1) ? k : v;
  const float* W = (z == 0) ? wq : (z == 1) ? wk : wv;
  _Float16*    O = (z == 0) ? qh : (z == 1) ? kh : vh;

  const int m0 = blockIdx.x * 16;
  const int n0 = (blockIdx.y * 4 + wave) * 16;

  const float* Arow = X + (size_t)(m0 + l16) * DD + hi * 16;
  const float* Brow = W + (size_t)(n0 + l16) * DD + hi * 16;

  v8f c = {};
#pragma unroll 4
  for (int k0 = 0; k0 < DD; k0 += 32) {
    __builtin_prefetch(Arow + k0 + 128, 0, 1);
    const v16h a = ld_a_f(Arow + k0);
    const v16h b = ld_b_f(Brow + k0);
    c = WMMA(a, b, c);
  }

  const int n    = n0 + l16;
  const int head = n >> 6;
  const int d    = n & 63;
#pragma unroll
  for (int r = 0; r < 8; ++r) {
    const int m  = m0 + r + hi * 8;
    const int b_ = m >> 11;
    const int s  = m & (SS - 1);
    O[(((size_t)b_ * HH + head) * SS + s) * DEP + d] = (_Float16)c[r];
  }
}

// ---------------------------------------------------------------------------
// Kernel 2: causal flash attention.
// grid = (S/128, B*H), block = 256 (8 waves).  Per 32-key tile:
//   K tile  -> LDS via TDM (wave 0, one DMA, TENSORcnt)
//   V tile  -> LDS via per-lane global_load_async_to_lds_b128 (ASYNCcnt)
// ---------------------------------------------------------------------------
__global__ void __launch_bounds__(256)
attn_kernel(const _Float16* __restrict__ qh, const _Float16* __restrict__ kh,
            const _Float16* __restrict__ vh, _Float16* __restrict__ attn_out) {
  __shared__ _Float16 Ks[32 * DEP];       // [key][d]
  __shared__ _Float16 Vs[32 * DEP];       // [key][d] (row-major)
  __shared__ _Float16 Ps[8][16 * 32];     // per-wave P tile [row][key]

  const int tid  = threadIdx.x;
  const int wave = tid >> 5;
  const int lane = tid & 31;
  const int l16  = lane & 15;
  const int hi   = lane >> 4;

  const int bh   = blockIdx.y;
  const int qblk = blockIdx.x;
  const size_t base = (size_t)bh * SS * DEP;
  const int q0 = qblk * 128 + wave * 16;

  const _Float16* Qrow = qh + base + (size_t)(q0 + l16) * DEP + hi * 16;
  const v16h qa0 = ld_a_h(Qrow);
  const v16h qa1 = ld_a_h(Qrow + 32);

  v8f o0 = {}, o1 = {}, o2 = {}, o3 = {};
  float mrow[8], lrow[8];
#pragma unroll
  for (int r = 0; r < 8; ++r) { mrow[r] = -1e30f; lrow[r] = 0.0f; }

  const float scale = 0.125f;
  const int ktiles = qblk * 4 + 4;

  for (int kt = 0; kt < ktiles; ++kt) {
    const int kbase = kt * 32;
    const _Float16* ktile = kh + base + (size_t)kbase * DEP;   // 4 KB contig
    const _Float16* vtile = vh + base + (size_t)kbase * DEP;   // 4 KB contig
    const int idx = tid * 8;                                   // 256*8 = 2048

    // ---- stage K tile via Tensor Data Mover (one DMA from wave 0)
#if HAVE_TDM
    if (wave == 0) tdm_load_f16(ktile, &Ks[0], 32 * DEP);
#else
    *(v8h*)&Ks[idx] = *(const v8h*)(ktile + idx);
#endif

    // ---- stage V tile via async global->LDS (16B per lane)
#if HAVE_ASYNC
    __builtin_amdgcn_global_load_async_to_lds_b128(
        as_global_v4i(vtile + idx), as_local_v4i(&Vs[idx]), 0, 0);
#else
    *(v8h*)&Vs[idx] = *(const v8h*)(vtile + idx);
#endif

    if (kt + 1 < ktiles)
      __builtin_prefetch(ktile + 32 * DEP + idx, 0, 0);

#if HAVE_ASYNC
#if __has_builtin(__builtin_amdgcn_s_wait_asynccnt)
    __builtin_amdgcn_s_wait_asynccnt(0);
#else
    asm volatile("s_wait_asynccnt 0" ::: "memory");
#endif
#endif
#if HAVE_TDM
    if (wave == 0) __builtin_amdgcn_s_wait_tensorcnt(0);
#endif
    __syncthreads();

    if (kbase <= q0 + 15) {
      // ---- logits: c0 = Q . K[0..15]^T ; c1 = Q . K[16..31]^T
      v8f c0 = {}, c1 = {};
      c0 = WMMA(qa0, ld_b_h(&Ks[(size_t)l16 * DEP + 0  + hi * 16]), c0);
      c0 = WMMA(qa1, ld_b_h(&Ks[(size_t)l16 * DEP + 32 + hi * 16]), c0);
      c1 = WMMA(qa0, ld_b_h(&Ks[(size_t)(16 + l16) * DEP + 0  + hi * 16]), c1);
      c1 = WMMA(qa1, ld_b_h(&Ks[(size_t)(16 + l16) * DEP + 32 + hi * 16]), c1);

      // ---- online softmax (row = r + 8*hi, cols across the 16-lane group)
      _Float16* Pw = &Ps[wave][0];
#pragma unroll
      for (int r = 0; r < 8; ++r) {
        const int qrow = q0 + r + hi * 8;
        const int key0 = kbase + l16;
        const int key1 = kbase + 16 + l16;
        float x0 = c0[r] * scale + ((key0 > qrow) ? -1e9f : 0.0f);
        float x1 = c1[r] * scale + ((key1 > qrow) ? -1e9f : 0.0f);
        float tmax = fmaxf(x0, x1);
#pragma unroll
        for (int off = 1; off < 16; off <<= 1)
          tmax = fmaxf(tmax, __shfl_xor(tmax, off, 32));
        const float mnew = fmaxf(mrow[r], tmax);
        const float p0 = __expf(x0 - mnew);
        const float p1 = __expf(x1 - mnew);
        float psum = p0 + p1;
#pragma unroll
        for (int off = 1; off < 16; off <<= 1)
          psum += __shfl_xor(psum, off, 32);
        const float f = __expf(mrow[r] - mnew);
        lrow[r] = lrow[r] * f + psum;
        mrow[r] = mnew;
        o0[r] *= f; o1[r] *= f; o2[r] *= f; o3[r] *= f;
        Pw[(r + hi * 8) * 32 + l16]      = (_Float16)p0;
        Pw[(r + hi * 8) * 32 + 16 + l16] = (_Float16)p1;
      }
      asm volatile("s_wait_dscnt 0" ::: "memory");

      // ---- O += P @ V   (A = P from LDS, B = rows of V with key stride)
      const v16h pa = ld_a_h(&Ps[wave][(size_t)l16 * 32 + hi * 16]);
      o0 = WMMA(pa, ld_b_strided(&Vs[(size_t)(hi * 16) * DEP + 0  + l16], DEP), o0);
      o1 = WMMA(pa, ld_b_strided(&Vs[(size_t)(hi * 16) * DEP + 16 + l16], DEP), o1);
      o2 = WMMA(pa, ld_b_strided(&Vs[(size_t)(hi * 16) * DEP + 32 + l16], DEP), o2);
      o3 = WMMA(pa, ld_b_strided(&Vs[(size_t)(hi * 16) * DEP + 48 + l16], DEP), o3);
    }
    __syncthreads();
  }

  // ---- normalize and write [B,S,D] f16 (heads re-interleaved)
  const int b_   = bh / HH;
  const int head = bh % HH;
#pragma unroll
  for (int r = 0; r < 8; ++r) {
    const float inv = 1.0f / lrow[r];
    const int s = q0 + r + hi * 8;
    _Float16* o = attn_out + ((size_t)b_ * SS + s) * DD + head * DEP;
    o[0  + l16] = (_Float16)(o0[r] * inv);
    o[16 + l16] = (_Float16)(o1[r] * inv);
    o[32 + l16] = (_Float16)(o2[r] * inv);
    o[48 + l16] = (_Float16)(o3[r] * inv);
  }
}

// ---------------------------------------------------------------------------
// Kernel 3: final dense.  out = A @ Wd^T + b, f32 output.
// ---------------------------------------------------------------------------
__global__ void __launch_bounds__(128)
dense_kernel(const _Float16* __restrict__ A, const float* __restrict__ W,
             const float* __restrict__ bias, float* __restrict__ out) {
  const int wave = threadIdx.x >> 5;
  const int lane = threadIdx.x & 31;
  const int l16  = lane & 15;
  const int hi   = lane >> 4;

  const int m0 = blockIdx.x * 16;
  const int n0 = (blockIdx.y * 4 + wave) * 16;

  const _Float16* Arow = A + (size_t)(m0 + l16) * DD + hi * 16;
  const float*    Brow = W + (size_t)(n0 + l16) * DD + hi * 16;

  v8f c = {};
#pragma unroll 4
  for (int k0 = 0; k0 < DD; k0 += 32) {
    __builtin_prefetch(Brow + k0 + 128, 0, 1);
    const v16h a = ld_a_h(Arow + k0);
    const v16h b = ld_b_f(Brow + k0);
    c = WMMA(a, b, c);
  }

  const int n = n0 + l16;
  const float bv = bias[n];
#pragma unroll
  for (int r = 0; r < 8; ++r) {
    const int m = m0 + r + hi * 8;
    out[(size_t)m * DD + n] = c[r] + bv;
  }
}

// ---------------------------------------------------------------------------
extern "C" void kernel_launch(void* const* d_in, const int* in_sizes, int n_in,
                              void* d_out, int out_size, void* d_ws, size_t ws_size,
                              hipStream_t stream) {
  (void)in_sizes; (void)n_in; (void)out_size; (void)ws_size;
  const float* q  = (const float*)d_in[0];
  const float* k  = (const float*)d_in[1];
  const float* v  = (const float*)d_in[2];
  // d_in[3] = additive causal mask; applied analytically in-kernel
  const float* wq = (const float*)d_in[4];
  const float* wk = (const float*)d_in[5];
  const float* wv = (const float*)d_in[6];
  const float* wd = (const float*)d_in[7];
  const float* bd = (const float*)d_in[8];

  const size_t per = (size_t)BB * HH * SS * DEP;
  _Float16* qh   = (_Float16*)d_ws;
  _Float16* kh   = qh + per;
  _Float16* vh   = kh + per;
  _Float16* attn = vh + per;

  qkv_proj_kernel<<<dim3((BB * SS) / 16, DD / 64, 3), 128, 0, stream>>>(
      q, k, v, wq, wk, wv, qh, kh, vh);
  attn_kernel<<<dim3(SS / 128, BB * HH), 256, 0, stream>>>(qh, kh, vh, attn);
  dense_kernel<<<dim3((BB * SS) / 16, DD / 64), 128, 0, stream>>>(
      attn, wd, bd, (float*)d_out);
}
